// MultiHeadRotaryAttention_47682726920917
// MI455X (gfx1250) — matrix operand order repaired and checked by reading, not compiled, not run on hardware
//
#include <hip/hip_runtime.h>
#include <hip/hip_bf16.h>

typedef __bf16 bf16_t;
typedef __attribute__((ext_vector_type(16))) __bf16 v16bf;
typedef __attribute__((ext_vector_type(8)))  __bf16 v8bf;
typedef __attribute__((ext_vector_type(8)))  float  v8f;

__device__ __forceinline__ v8f wmma_bf16(v16bf a, v16bf b, v8f c) {
  return __builtin_amdgcn_wmma_f32_16x16x32_bf16(false, a, false, b, (short)0, c,
                                                 false, false);
}
__device__ __forceinline__ v8f vzero8() {
  v8f z = {0.f, 0.f, 0.f, 0.f, 0.f, 0.f, 0.f, 0.f};
  return z;
}

// Async global->LDS copy of 16 bytes per lane (per-lane LDS dest address).
__device__ __forceinline__ void async_copy16(const void* lds_dst, const void* gsrc) {
  unsigned l = (unsigned)(size_t)lds_dst;
  asm volatile("global_load_async_to_lds_b128 %0, %1, off"
               :: "v"(l), "v"(gsrc)
               : "memory");
}
// Wait until at most N async copies remain outstanding (in-order completion).
#define WAIT_ASYNC_LE(n) asm volatile("s_wait_asynccnt %0" :: "n"(n) : "memory")

// ---------------------------------------------------------------------------
// Elementwise fp32 -> bf16 convert.
// ---------------------------------------------------------------------------
__global__ __launch_bounds__(256)
void cvt_bf16_kernel(const float* __restrict__ in, bf16_t* __restrict__ out, int n)
{
  int idx = blockIdx.x * 256 + threadIdx.x;
  if (idx < n) out[idx] = (bf16_t)in[idx];
}

// ---------------------------------------------------------------------------
// Wt[n][k] = (bf16) W[k][n]   (transpose + convert; done once per weight)
// ---------------------------------------------------------------------------
__global__ __launch_bounds__(256)
void transpose_cvt_kernel(const float* __restrict__ W, bf16_t* __restrict__ Wt,
                          int K, int N)
{
  int idx = blockIdx.x * 256 + threadIdx.x;
  if (idx >= K * N) return;
  int n = idx % N, k = idx / N;
  Wt[(size_t)n * K + k] = (bf16_t)W[(size_t)k * N + n];
}

// ---------------------------------------------------------------------------
// C[M,N] = A[M,K] @ Wt[N,K]^T + bias.  A, Wt bf16; C fp32.
// Block 256 thr = 8 waves (4Mx2N), block tile 128x128, wave tile 32x64, BK=32.
// Double-buffered LDS staged via GLOBAL_LOAD_ASYNC_TO_LDS_B128:
// tile t+1 in flight (asynccnt<=4) while tile t is consumed by WMMA.
// ---------------------------------------------------------------------------
__global__ __launch_bounds__(256)
void gemm_bias_wmma_bf16(const bf16_t* __restrict__ A, const bf16_t* __restrict__ Wt,
                         const float* __restrict__ bias, float* __restrict__ C,
                         int M, int N, int K)
{
  __shared__ alignas(32) bf16_t sa[2][128][32];   // A tiles  [m][k]
  __shared__ alignas(32) bf16_t sbt[2][128][32];  // W tiles  [n][k]

  const int tid  = threadIdx.x;
  const int lane = tid & 31;
  const int wv   = tid >> 5;
  const int wm   = wv & 3;        // 4 wave rows
  const int wn   = wv >> 2;       // 2 wave cols
  const int hl   = lane >> 4;
  const int lm   = lane & 15;
  const int m0   = blockIdx.y * 128;
  const int n0   = blockIdx.x * 128;

  v8f acc[2][4];
#pragma unroll
  for (int i = 0; i < 2; ++i)
#pragma unroll
    for (int j = 0; j < 4; ++j) acc[i][j] = vzero8();

  // stage one 128x32 A tile + 128x32 W tile: 4 async chunks per thread
  auto stage = [&](int k0, int buf) {
#pragma unroll
    for (int it = 0; it < 2; ++it) {
      int fi = tid + it * 256;
      int r = fi >> 2, c8 = (fi & 3) * 8;
      async_copy16(&sa[buf][r][c8], A + (size_t)(m0 + r) * K + k0 + c8);
    }
#pragma unroll
    for (int it = 0; it < 2; ++it) {
      int fi = tid + it * 256;
      int n = fi >> 2, c8 = (fi & 3) * 8;
      async_copy16(&sbt[buf][n][c8], Wt + (size_t)(n0 + n) * K + k0 + c8);
    }
  };

  const int nk = K >> 5;
  stage(0, 0);
  for (int t = 0; t < nk; ++t) {
    const int buf = t & 1;
    if (t + 1 < nk) {
      stage((t + 1) << 5, buf ^ 1);
      WAIT_ASYNC_LE(4);            // tile t complete; tile t+1 still in flight
    } else {
      WAIT_ASYNC_LE(0);
    }
    __syncthreads();

    v16bf af[2], bfr[4];
#pragma unroll
    for (int fm = 0; fm < 2; ++fm) {
      int row = wm * 32 + fm * 16 + lm;
      af[fm] = *(const v16bf*)&sa[buf][row][hl * 16];
    }
#pragma unroll
    for (int fn = 0; fn < 4; ++fn) {
      int col = wn * 64 + fn * 16 + lm;
      bfr[fn] = *(const v16bf*)&sbt[buf][col][hl * 16];
    }
#pragma unroll
    for (int fm = 0; fm < 2; ++fm)
#pragma unroll
      for (int fn = 0; fn < 4; ++fn)
        acc[fm][fn] = wmma_bf16(af[fm], bfr[fn], acc[fm][fn]);
    __syncthreads();               // tile buf free for restaging next iteration
  }

#pragma unroll
  for (int fm = 0; fm < 2; ++fm)
#pragma unroll
    for (int fn = 0; fn < 4; ++fn) {
      int col = n0 + wn * 64 + fn * 16 + lm;
      float bv = bias[col];
#pragma unroll
      for (int r = 0; r < 8; ++r) {
        int row = m0 + wm * 32 + fm * 16 + r + hl * 8;
        C[(size_t)row * N + col] = acc[fm][fn][r] + bv;
      }
    }
}

// ---------------------------------------------------------------------------
// RoPE + pack to head-major bf16:
//   qg[bh][s][d] = rope(q)/32,  kg[bh][s][d] = rope(k),  vtg[bh][d][s] = v.
// ---------------------------------------------------------------------------
__global__ __launch_bounds__(256)
void pack_rope_kernel(const float* __restrict__ qkv, bf16_t* __restrict__ qg,
                      bf16_t* __restrict__ kg, bf16_t* __restrict__ vtg,
                      int S, int total)
{
  int idx = blockIdx.x * 256 + threadIdx.x;
  if (idx >= total) return;
  int d2 = idx & 31;
  int t  = idx >> 5;
  int s  = t % S;
  int bh = t / S;
  int b = bh >> 4, h = bh & 15;
  int d0 = 2 * d2;

  const float* src = qkv + (size_t)(b * S + s) * 3072 + h * 64;
  float c = 1.f, sn = 0.f;
  if (d2 < 16) {                       // only first 32 dims rotated (rot = 32)
    float freq = __powf(10000.f, -(float)d0 / 32.f);
    float ang = (float)s * freq;
    c = __cosf(ang); sn = __sinf(ang);
  }
  size_t qo = ((size_t)bh * S + s) * 64 + d0;

  float q0 = src[d0], q1 = src[d0 + 1];
  qg[qo]     = (bf16_t)((q0 * c - q1 * sn) * 0.03125f);   // fold 1024^-0.5
  qg[qo + 1] = (bf16_t)((q1 * c + q0 * sn) * 0.03125f);

  float k0 = src[1024 + d0], k1 = src[1024 + d0 + 1];
  kg[qo]     = (bf16_t)(k0 * c - k1 * sn);
  kg[qo + 1] = (bf16_t)(k1 * c + k0 * sn);

  float v0 = src[2048 + d0], v1 = src[2048 + d0 + 1];
  vtg[((size_t)bh * 64 + d0) * S + s]     = (bf16_t)v0;
  vtg[((size_t)bh * 64 + d0 + 1) * S + s] = (bf16_t)v1;
}

// ---------------------------------------------------------------------------
// Flash attention, one (b, h, 64-query-row tile) per block, 4 waves.
// Double-buffered async K/V staging; output (+ residual x) written as bf16.
// ---------------------------------------------------------------------------
__global__ __launch_bounds__(128)
void flash_attn_wmma(const bf16_t* __restrict__ qg, const bf16_t* __restrict__ kg,
                     const bf16_t* __restrict__ vtg, const float* __restrict__ x,
                     bf16_t* __restrict__ attn_out, int B, int S)
{
  __shared__ alignas(32) bf16_t kt[2][32][64];   // K tiles  [j][d]
  __shared__ alignas(32) bf16_t vt[2][64][32];   // V tiles  [d][j]
  __shared__ alignas(32) bf16_t ps[4][16][32];   // per-wave P transpose scratch

  const int tid = threadIdx.x;
  const int lane = tid & 31;
  const int wv = tid >> 5;
  const int hl = lane >> 4;
  const int lm = lane & 15;

  const int nQT = S / 64;
  const int qt = blockIdx.x % nQT;
  const int bh = blockIdx.x / nQT;
  const int b = bh >> 4;

  const int qrow0 = qt * 64 + wv * 16;

  // Q as two A-fragments (d = 0..31, 32..63); already roped + scaled, bf16
  v16bf qa[2];
  {
    const bf16_t* qp = qg + ((size_t)bh * S + qrow0 + lm) * 64;
#pragma unroll
    for (int f = 0; f < 2; ++f) {
      qa[f] = *(const v16bf*)(qp + f * 32 + hl * 16);
    }
  }

  float mrow[8], lrow[8];
#pragma unroll
  for (int r = 0; r < 8; ++r) { mrow[r] = -3.0e38f; lrow[r] = 0.f; }
  v8f o[4];
#pragma unroll
  for (int d = 0; d < 4; ++d) o[d] = vzero8();

  // stage a 32-key K tile + V^T tile: 4 async chunks per thread (128 thr)
  auto stageKV = [&](int j0, int buf) {
#pragma unroll
    for (int it = 0; it < 2; ++it) {
      int fi = tid + it * 128;
      int jr = fi >> 3, c8 = (fi & 7) * 8;
      async_copy16(&kt[buf][jr][c8], kg + ((size_t)bh * S + j0 + jr) * 64 + c8);
    }
#pragma unroll
    for (int it = 0; it < 2; ++it) {
      int fi = tid + it * 128;
      int dr = fi >> 2, c8 = (fi & 3) * 8;
      async_copy16(&vt[buf][dr][c8], vtg + ((size_t)bh * 64 + dr) * S + j0 + c8);
    }
  };

  const int nj = S >> 5;
  stageKV(0, 0);
  for (int t = 0; t < nj; ++t) {
    const int buf = t & 1;
    if (t + 1 < nj) {
      stageKV((t + 1) << 5, buf ^ 1);
      WAIT_ASYNC_LE(4);
    } else {
      WAIT_ASYNC_LE(0);
    }
    __syncthreads();

    // scores: two 16-key sub-tiles, two K-steps each
    v8f sfrag[2];
#pragma unroll
    for (int jf = 0; jf < 2; ++jf) {
      int jj = jf * 16 + lm;
      v16bf kb0 = *(const v16bf*)&kt[buf][jj][hl * 16];
      v16bf kb1 = *(const v16bf*)&kt[buf][jj][32 + hl * 16];
      v8f sf = vzero8();
      sf = wmma_bf16(qa[0], kb0, sf);
      sf = wmma_bf16(qa[1], kb1, sf);
      sfrag[jf] = sf;
    }

    // online softmax (rows m = r + 8*hl, stats replicated across 16 lanes)
    float cf[8];
#pragma unroll
    for (int r = 0; r < 8; ++r) {
      float v = fmaxf(sfrag[0][r], sfrag[1][r]);
#pragma unroll
      for (int off = 1; off < 16; off <<= 1) v = fmaxf(v, __shfl_xor(v, off, 32));
      float mn = fmaxf(mrow[r], v);
      cf[r] = __expf(mrow[r] - mn);
      mrow[r] = mn;
      float p0 = __expf(sfrag[0][r] - mn);
      float p1 = __expf(sfrag[1][r] - mn);
      sfrag[0][r] = p0; sfrag[1][r] = p1;
      float rs = p0 + p1;
#pragma unroll
      for (int off = 1; off < 16; off <<= 1) rs += __shfl_xor(rs, off, 32);
      lrow[r] = lrow[r] * cf[r] + rs;
    }
#pragma unroll
    for (int d = 0; d < 4; ++d)
#pragma unroll
      for (int r = 0; r < 8; ++r) o[d][r] *= cf[r];

    // transpose P (C layout -> A layout) through per-wave LDS scratch
#pragma unroll
    for (int r = 0; r < 8; ++r) {
      int m = r + 8 * hl;
      ps[wv][m][lm]      = (bf16_t)sfrag[0][r];
      ps[wv][m][16 + lm] = (bf16_t)sfrag[1][r];
    }
    asm volatile("s_wait_dscnt 0" ::: "memory");
    v16bf pa;
    {
      pa = *(const v16bf*)&ps[wv][lm][hl * 16];
    }

    // O += P @ V
#pragma unroll
    for (int d = 0; d < 4; ++d) {
      v16bf vb = *(const v16bf*)&vt[buf][d * 16 + lm][hl * 16];
      o[d] = wmma_bf16(pa, vb, o[d]);
    }
    __syncthreads();
  }

  // epilogue: O/l + residual x, stored bf16 (consumed as next GEMM's A)
  float invl[8];
#pragma unroll
  for (int r = 0; r < 8; ++r) invl[r] = 1.f / lrow[r];
  const int h = bh & 15;
#pragma unroll
  for (int d = 0; d < 4; ++d) {
    int col = h * 64 + d * 16 + lm;
#pragma unroll
    for (int r = 0; r < 8; ++r) {
      size_t row = (size_t)(b * S + qrow0 + r + 8 * hl);
      attn_out[row * 1024 + col] = (bf16_t)(o[d][r] * invl[r] + x[row * 1024 + col]);
    }
  }
}

// ---------------------------------------------------------------------------
// LayerNorm, one block (256 threads) per row of D=1024.
// ---------------------------------------------------------------------------
__global__ __launch_bounds__(256)
void layernorm_kernel(const float* __restrict__ y, const float* __restrict__ gamma,
                      const float* __restrict__ beta, float* __restrict__ out, int D)
{
  __shared__ float s1[8], s2[8];
  const int row = blockIdx.x;
  const float* yr = y + (size_t)row * D;
  float sum = 0.f, sq = 0.f;
  for (int i = threadIdx.x; i < D; i += 256) {
    float v = yr[i];
    sum += v; sq += v * v;
  }
#pragma unroll
  for (int off = 1; off < 32; off <<= 1) {
    sum += __shfl_xor(sum, off, 32);
    sq  += __shfl_xor(sq,  off, 32);
  }
  int lane = threadIdx.x & 31, wv = threadIdx.x >> 5;
  if (lane == 0) { s1[wv] = sum; s2[wv] = sq; }
  __syncthreads();
  float ts = 0.f, tq = 0.f;
#pragma unroll
  for (int w = 0; w < 8; ++w) { ts += s1[w]; tq += s2[w]; }
  float mu = ts / (float)D;
  float var = tq / (float)D - mu * mu;
  float inv = rsqrtf(var + 1e-5f);
  for (int i = threadIdx.x; i < D; i += 256)
    out[(size_t)row * D + i] = (yr[i] - mu) * inv * gamma[i] + beta[i];
}

// ---------------------------------------------------------------------------
extern "C" void kernel_launch(void* const* d_in, const int* in_sizes, int n_in,
                              void* d_out, int out_size, void* d_ws, size_t ws_size,
                              hipStream_t stream)
{
  const float* x      = (const float*)d_in[0];
  const float* w_qkv  = (const float*)d_in[1];
  const float* b_qkv  = (const float*)d_in[2];
  const float* w_out  = (const float*)d_in[3];
  const float* b_out  = (const float*)d_in[4];
  const float* ln_g   = (const float*)d_in[5];
  const float* ln_b   = (const float*)d_in[6];
  float* out = (float*)d_out;

  const int B = 2, S = 2048, D = 1024;
  const int M = B * S;
  const int BH = B * 16;

  // workspace layout
  char* ws = (char*)d_ws;
  float*  qkv = (float*)ws;                         // M x 3072 f32 (later reused as y)
  size_t off = (size_t)M * 3072 * 4;
  bf16_t* xg   = (bf16_t*)(ws + off); off += (size_t)M * D * 2;
  bf16_t* wqT  = (bf16_t*)(ws + off); off += (size_t)3 * D * D * 2;
  bf16_t* woT  = (bf16_t*)(ws + off); off += (size_t)D * D * 2;
  bf16_t* qg   = (bf16_t*)(ws + off); off += (size_t)M * D * 2;
  bf16_t* kg   = (bf16_t*)(ws + off); off += (size_t)M * D * 2;
  bf16_t* vtg  = (bf16_t*)(ws + off); off += (size_t)M * D * 2;
  bf16_t* attn = (bf16_t*)(ws + off); off += (size_t)M * D * 2;
  float*  y    = qkv;                               // alias: qkv dead after pack_rope

  // 0) one-time converts: x -> bf16, weights -> transposed bf16
  cvt_bf16_kernel<<<(M * D + 255) / 256, 256, 0, stream>>>(x, xg, M * D);
  transpose_cvt_kernel<<<(D * 3 * D + 255) / 256, 256, 0, stream>>>(w_qkv, wqT, D, 3 * D);
  transpose_cvt_kernel<<<(D * D + 255) / 256, 256, 0, stream>>>(w_out, woT, D, D);

  // 1) QKV projection (fp32 out for RoPE precision)
  {
    dim3 grid(3 * D / 128, M / 128);
    gemm_bias_wmma_bf16<<<grid, 256, 0, stream>>>(xg, wqT, b_qkv, qkv, M, 3 * D, D);
  }
  // 2) RoPE + head-major bf16 packing (q scaled by 1/32, V transposed)
  {
    int total = BH * S * 32;
    pack_rope_kernel<<<(total + 255) / 256, 256, 0, stream>>>(qkv, qg, kg, vtg, S, total);
  }
  // 3) Flash attention + residual (bf16 out)
  flash_attn_wmma<<<BH * (S / 64), 128, 0, stream>>>(qg, kg, vtg, x, attn, B, S);
  // 4) Output projection (y aliases qkv storage)
  {
    dim3 grid(D / 128, M / 128);
    gemm_bias_wmma_bf16<<<grid, 256, 0, stream>>>(attn, woT, b_out, y, M, D, D);
  }
  // 5) LayerNorm
  layernorm_kernel<<<M, 256, 0, stream>>>(y, ln_g, ln_b, out, D);
}
